// PointNetFeaturePropagation_84756884619599
// MI455X (gfx1250) — matrix operand order repaired and checked by reading, not compiled, not run on hardware
//
#include <hip/hip_runtime.h>
#include <hip/hip_bf16.h>

// ---------------------------------------------------------------------------
// PointNet Feature Propagation for MI455X (gfx1250, wave32, WMMA)
// Pipeline: knn3 -> build_npts(f16) -> wmma GEMM1 -> stats -> finalize ->
//           bn_relu->f16 -> wmma GEMM2 -> stats -> finalize -> bn_relu+transpose
// ---------------------------------------------------------------------------

typedef __attribute__((ext_vector_type(16))) _Float16 v16h;
typedef __attribute__((ext_vector_type(8)))  _Float16 v8h;
typedef __attribute__((ext_vector_type(4)))  _Float16 v4h;
typedef __attribute__((ext_vector_type(8)))  float    v8f;
typedef int vsi4 __attribute__((vector_size(16)));   // matches async-LDS builtin

static constexpr int B_   = 8;
static constexpr int N_   = 8192;
static constexpr int S_   = 2048;
static constexpr int D1_  = 128;
static constexpr int D2_  = 256;
static constexpr int CIN_ = 384;   // D1 + D2
static constexpr int C1_  = 256;
static constexpr int C2_  = 128;
static constexpr long NP_ = (long)B_ * N_;   // 65536 total points

#if defined(__has_builtin)
#if __has_builtin(__builtin_amdgcn_global_load_async_to_lds_b128) && \
    __has_builtin(__builtin_amdgcn_s_wait_asynccnt)
#define USE_ASYNC_LDS 1
#endif
#endif
#ifndef USE_ASYNC_LDS
#define USE_ASYNC_LDS 0
#endif

__device__ __forceinline__ v8f wmma_f16(v16h a, v16h b, v8f c) {
  // (neg_a, A, neg_b, B, c_mod, C, reuse_a, reuse_b)
  return __builtin_amdgcn_wmma_f32_16x16x32_f16(false, a, false, b, (short)0, c,
                                                false, false);
}
__device__ __forceinline__ v16h cat8(v8h a, v8h b) {
  return __builtin_shufflevector(a, b, 0,1,2,3,4,5,6,7,8,9,10,11,12,13,14,15);
}

// ---------------------------------------------------------------------------
// Kernel 0: convert w1 [256x384] and w2 [128x256] fp32 -> f16
// ---------------------------------------------------------------------------
__global__ __launch_bounds__(256) void cvt_weights(
    const float* __restrict__ w1, const float* __restrict__ w2,
    _Float16* __restrict__ w1h, _Float16* __restrict__ w2h) {
  int i = blockIdx.x * 256 + threadIdx.x;
  if (i < C1_ * CIN_) w1h[i] = (_Float16)w1[i];
  if (i < C2_ * C1_)  w2h[i] = (_Float16)w2[i];
}

// ---------------------------------------------------------------------------
// Kernel 1: 3-NN search. One thread per query point; xyz2 of the batch cached
// in LDS (SoA, 24KB).
// Packed-key top-3: key = (float_bits(d) & 0xFFFFF800) | s. d >= 0 so float
// bits order as u32; 11 low bits carry the index (S=2048); truncation error
// 2^-12 relative, far below the f16 GEMM noise. Sorted insert is a pure
// min/max network (5 VALU ops, fusable to v_med3/v_max3), no cndmask chains.
// grid = (N/256, B), block = 256
// ---------------------------------------------------------------------------
__global__ __launch_bounds__(256) void knn3(
    const float* __restrict__ xyz1, const float* __restrict__ xyz2,
    int* __restrict__ idx, float* __restrict__ wts) {
  __shared__ float sx[S_], sy[S_], sz[S_];
  const int b = blockIdx.y;
  const int n = blockIdx.x * 256 + threadIdx.x;
  const float* p2 = xyz2 + (long)b * S_ * 3;
  for (int i = threadIdx.x; i < S_; i += 256) {
    sx[i] = p2[i * 3 + 0];
    sy[i] = p2[i * 3 + 1];
    sz[i] = p2[i * 3 + 2];
  }
  __syncthreads();

  const long g = (long)b * N_ + n;
  const float x = xyz1[g * 3 + 0], y = xyz1[g * 3 + 1], z = xyz1[g * 3 + 2];

  unsigned k0 = 0xFFFFFFFFu, k1 = 0xFFFFFFFFu, k2 = 0xFFFFFFFFu;
#pragma unroll 4
  for (int s = 0; s < S_; ++s) {
    const float dx = x - sx[s], dy = y - sy[s], dz = z - sz[s];
    const float d = dx * dx + dy * dy + dz * dz;   // == max(d,0) of reference
    const unsigned k = (__float_as_uint(d) & 0xFFFFF800u) | (unsigned)s;
    const unsigned m0 = k0 > k ? k0 : k;           // max(k0,k)
    k0 = k0 < k ? k0 : k;                          // min(k0,k)
    const unsigned m1 = k1 > m0 ? k1 : m0;         // max3(k0,k,k1)
    k1 = k1 < m0 ? k1 : m0;                        // med3(k0,k,k1)
    k2 = k2 < m1 ? k2 : m1;                        // 3rd smallest
  }
  const int   i0 = (int)(k0 & 0x7FFu);
  const int   i1 = (int)(k1 & 0x7FFu);
  const int   i2 = (int)(k2 & 0x7FFu);
  const float d0 = __uint_as_float(k0 & 0xFFFFF800u);
  const float d1 = __uint_as_float(k1 & 0xFFFFF800u);
  const float d2 = __uint_as_float(k2 & 0xFFFFF800u);
  const float r0 = 1.f / (d0 + 1e-8f);
  const float r1 = 1.f / (d1 + 1e-8f);
  const float r2 = 1.f / (d2 + 1e-8f);
  const float rs = 1.f / (r0 + r1 + r2);
  idx[g * 3 + 0] = i0;  wts[g * 3 + 0] = r0 * rs;
  idx[g * 3 + 1] = i1;  wts[g * 3 + 1] = r1 * rs;
  idx[g * 3 + 2] = i2;  wts[g * 3 + 2] = r2 * rs;
}

// ---------------------------------------------------------------------------
// Kernel 2: build new_points f16 [NP][384] = concat(points1 f16, interp f16).
// One wave per point at a time (8 points per wave, 64 per block); a wave reads
// the three 1KB neighbor rows coalesced and blends in registers.
// ---------------------------------------------------------------------------
__global__ __launch_bounds__(256) void build_npts(
    const float* __restrict__ points1, const float* __restrict__ points2,
    const int* __restrict__ idx, const float* __restrict__ wts,
    _Float16* __restrict__ npts) {
  const int wave = threadIdx.x >> 5;
  const int lane = threadIdx.x & 31;
  const long pbase = (long)blockIdx.x * 64 + (long)wave * 8;
  for (int pp = 0; pp < 8; ++pp) {
    const long n = pbase + pp;
    const int  b = (int)(n >> 13);
    const int  j0 = idx[n * 3 + 0], j1 = idx[n * 3 + 1], j2 = idx[n * 3 + 2];
    const float w0 = wts[n * 3 + 0], w1 = wts[n * 3 + 1], w2 = wts[n * 3 + 2];
    const float* P2 = points2 + (long)b * S_ * D2_;

    // interpolated 256 channels: lane covers c = lane*8 .. lane*8+7
    const float* r0 = P2 + (long)j0 * D2_ + lane * 8;
    const float* r1 = P2 + (long)j1 * D2_ + lane * 8;
    const float* r2 = P2 + (long)j2 * D2_ + lane * 8;
    float4 a0 = *(const float4*)r0,      a1 = *(const float4*)(r0 + 4);
    float4 b0 = *(const float4*)r1,      b1 = *(const float4*)(r1 + 4);
    float4 c0 = *(const float4*)r2,      c1 = *(const float4*)(r2 + 4);
    v8h o;
    o[0] = (_Float16)(w0 * a0.x + w1 * b0.x + w2 * c0.x);
    o[1] = (_Float16)(w0 * a0.y + w1 * b0.y + w2 * c0.y);
    o[2] = (_Float16)(w0 * a0.z + w1 * b0.z + w2 * c0.z);
    o[3] = (_Float16)(w0 * a0.w + w1 * b0.w + w2 * c0.w);
    o[4] = (_Float16)(w0 * a1.x + w1 * b1.x + w2 * c1.x);
    o[5] = (_Float16)(w0 * a1.y + w1 * b1.y + w2 * c1.y);
    o[6] = (_Float16)(w0 * a1.z + w1 * b1.z + w2 * c1.z);
    o[7] = (_Float16)(w0 * a1.w + w1 * b1.w + w2 * c1.w);
    *(v8h*)(npts + n * CIN_ + D1_ + lane * 8) = o;

    // points1 128 channels: lane covers c = lane*4 .. lane*4+3
    float4 v = *(const float4*)(points1 + n * D1_ + lane * 4);
    v4h h;
    h[0] = (_Float16)v.x; h[1] = (_Float16)v.y;
    h[2] = (_Float16)v.z; h[3] = (_Float16)v.w;
    *(v4h*)(npts + n * CIN_ + lane * 4) = h;
  }
}

// ---------------------------------------------------------------------------
// WMMA GEMM: Y[n][m] = sum_k W[m][k] * X[n][k]   (f16 in, f32 out)
// Block = 8 waves, 64 points per block, all M channels per block.
// X tile staged in padded LDS (async-to-LDS when available); W fragments
// streamed from L2. A and B fragments both use contiguous k = 16*hi + s (16x16x32 f16).
// ---------------------------------------------------------------------------
template <int M, int K>
__global__ __launch_bounds__(256) void gemm_wmma(
    const _Float16* __restrict__ W,   // [M][K] row-major
    const _Float16* __restrict__ X,   // [NP][K] row-major (point-major)
    float* __restrict__ Y) {          // [NP][M]
  constexpr int NPTS = 64;
  constexpr int KT   = K / 32;
  constexpr int MTPW = M / 16 / 8;       // m-tiles per wave (2 or 1)
  constexpr int ROWH = K + 8;            // padded LDS row (halfs)
  __shared__ alignas(16) _Float16 xs[NPTS * ROWH];

  const int tid  = threadIdx.x;
  const int wave = tid >> 5;
  const int lane = tid & 31;
  const int ll   = lane & 15;            // row/col within 16-tile
  const int hi   = lane >> 4;            // lane-half selector
  const long nbase = (long)blockIdx.x * NPTS;

  // cooperative, coalesced load of the 64xK f16 tile into LDS
  constexpr int CH = K / 8;              // 16-byte chunks per row
#if USE_ASYNC_LDS
  for (int i = tid; i < NPTS * CH; i += 256) {
    const int r = i / CH, c = i - r * CH;
    const _Float16* gp = X + (nbase + r) * K + c * 8;
    _Float16*       lp = xs + r * ROWH + c * 8;
    __builtin_amdgcn_global_load_async_to_lds_b128(
        (__attribute__((address_space(1))) vsi4*)gp,
        (__attribute__((address_space(3))) vsi4*)lp,
        0, 0);
  }
  __builtin_amdgcn_s_wait_asynccnt(0);
#else
  for (int i = tid; i < NPTS * CH; i += 256) {
    const int r = i / CH, c = i - r * CH;
    *(float4*)(xs + r * ROWH + c * 8) =
        *(const float4*)(X + (nbase + r) * K + c * 8);
  }
#endif
  __syncthreads();

  v8f acc[MTPW][4];
#pragma unroll
  for (int mt = 0; mt < MTPW; ++mt)
#pragma unroll
    for (int nt = 0; nt < 4; ++nt) acc[mt][nt] = v8f{};

  for (int kt = 0; kt < KT; ++kt) {
    const int k0 = kt * 32;
    // B fragments: col n = nt*16+ll, 16 contiguous K at k0 + 16*hi
    v16h bf[4];
#pragma unroll
    for (int nt = 0; nt < 4; ++nt) {
      const _Float16* p = xs + (nt * 16 + ll) * ROWH + k0 + hi * 16;
      bf[nt] = cat8(*(const v8h*)p, *(const v8h*)(p + 8));
    }
    // A fragments: row m, 16 contiguous K at k0 + 16*hi
#pragma unroll
    for (int mt = 0; mt < MTPW; ++mt) {
      const int m = (wave * MTPW + mt) * 16 + ll;
      const _Float16* wp = W + (long)m * K + k0 + hi * 16;
      const v16h af = cat8(*(const v8h*)wp, *(const v8h*)(wp + 8));
#pragma unroll
      for (int nt = 0; nt < 4; ++nt) acc[mt][nt] = wmma_f16(af, bf[nt], acc[mt][nt]);
    }
  }

  // C layout: vgpr r, lane -> (m = r + 8*hi, n = ll)
#pragma unroll
  for (int mt = 0; mt < MTPW; ++mt) {
    const int mb = (wave * MTPW + mt) * 16 + hi * 8;
#pragma unroll
    for (int nt = 0; nt < 4; ++nt) {
      float* yp = Y + (nbase + nt * 16 + ll) * M + mb;
#pragma unroll
      for (int r = 0; r < 8; ++r) yp[r] = acc[mt][nt][r];
    }
  }
}

// ---------------------------------------------------------------------------
// Column statistics: per-channel sum and sum-of-squares over all NP points.
// sums[0..C) = sum, sums[C..2C) = sumsq (must be pre-zeroed).
// ---------------------------------------------------------------------------
template <int C>
__global__ __launch_bounds__(256) void col_stats(
    const float* __restrict__ Y, float* __restrict__ sums) {
  constexpr int PPB = 256 / C;
  const int c  = threadIdx.x % C;
  const int pr = threadIdx.x / C;
  const long p0 = (long)blockIdx.x * 128;
  float s = 0.f, q = 0.f;
  for (int p = pr; p < 128; p += PPB) {
    const float v = Y[(p0 + p) * C + c];
    s += v; q += v * v;
  }
  __shared__ float ls[256], lq[256];
  ls[threadIdx.x] = s; lq[threadIdx.x] = q;
  __syncthreads();
  if (pr == 0) {
#pragma unroll
    for (int j = 1; j < PPB; ++j) { s += ls[c + j * C]; q += lq[c + j * C]; }
    atomicAdd(&sums[c], s);
    atomicAdd(&sums[C + c], q);
  }
}

// finalize: scale = g*rsqrt(var+eps), shift = beta - mean*scale.
// (BN cancels the conv bias exactly, so b1/b2 are dropped.)
template <int C>
__global__ void bn_finalize(const float* __restrict__ sums,
                            const float* __restrict__ g,
                            const float* __restrict__ beta,
                            float* __restrict__ sc, float* __restrict__ sh) {
  const int c = threadIdx.x;
  if (c >= C) return;
  const float inv = 1.f / (float)NP_;
  const float mean = sums[c] * inv;
  const float var  = sums[C + c] * inv - mean * mean;
  const float s = g[c] * rsqrtf(var + 1e-5f);
  sc[c] = s;
  sh[c] = beta[c] - mean * s;
}

// BN+ReLU and convert to f16 (layer-1 activations, point-major [NP][256])
__global__ __launch_bounds__(256) void bn_relu_h(
    const float* __restrict__ Y, const float* __restrict__ sc,
    const float* __restrict__ sh, _Float16* __restrict__ Xh) {
  const long i = (long)blockIdx.x * 256 + threadIdx.x;   // NP*256 elements
  const int c = (int)(i & 255);
  const float v = fmaxf(Y[i] * sc[c] + sh[c], 0.f);
  Xh[i] = (_Float16)v;
}

// BN+ReLU + transpose [NP][128] -> out [B][128][N] via LDS tile
__global__ __launch_bounds__(256) void bn_relu_out(
    const float* __restrict__ Y, const float* __restrict__ sc,
    const float* __restrict__ sh, float* __restrict__ out) {
  __shared__ float t[64][C2_ + 1];
  const long nbase = (long)blockIdx.x * 64;
  for (int i = threadIdx.x; i < 64 * C2_; i += 256) {
    const int p = i >> 7, c = i & (C2_ - 1);
    const float v = Y[(nbase + p) * C2_ + c];
    t[p][c] = fmaxf(v * sc[c] + sh[c], 0.f);
  }
  __syncthreads();
  const int  b  = (int)(nbase >> 13);
  const long nl = nbase & (N_ - 1);
  for (int i = threadIdx.x; i < 64 * C2_; i += 256) {
    const int c = i >> 6, p = i & 63;           // p fastest -> coalesced
    out[((long)b * C2_ + c) * N_ + nl + p] = t[p][c];
  }
}

// ---------------------------------------------------------------------------
// Host-side orchestration (graph-capture safe: only kernel launches +
// hipMemsetAsync on `stream`).
// ---------------------------------------------------------------------------
static inline size_t align256(size_t x) { return (x + 255) & ~(size_t)255; }

extern "C" void kernel_launch(void* const* d_in, const int* in_sizes, int n_in,
                              void* d_out, int out_size, void* d_ws, size_t ws_size,
                              hipStream_t stream) {
  const float* xyz1    = (const float*)d_in[0];
  const float* xyz2    = (const float*)d_in[1];
  const float* points1 = (const float*)d_in[2];
  const float* points2 = (const float*)d_in[3];
  const float* w1      = (const float*)d_in[4];
  const float* g1      = (const float*)d_in[6];
  const float* beta1   = (const float*)d_in[7];
  const float* w2      = (const float*)d_in[8];
  const float* g2      = (const float*)d_in[10];
  const float* beta2   = (const float*)d_in[11];
  float* out = (float*)d_out;

  // workspace carve-up
  char* ws = (char*)d_ws;
  size_t off = 0;
  _Float16* w1h  = (_Float16*)(ws + off); off = align256(off + (size_t)C1_ * CIN_ * 2);
  _Float16* w2h  = (_Float16*)(ws + off); off = align256(off + (size_t)C2_ * C1_ * 2);
  float*    stats= (float*)   (ws + off); size_t stats_off = off;
  off = align256(off + 4 * 256 * sizeof(float));          // sum1,sq1,sum2,sq2
  float*    sc1  = (float*)(ws + off); off = align256(off + C1_ * 4);
  float*    sh1  = (float*)(ws + off); off = align256(off + C1_ * 4);
  float*    sc2  = (float*)(ws + off); off = align256(off + C2_ * 4);
  float*    sh2  = (float*)(ws + off); off = align256(off + C2_ * 4);
  int*      idx  = (int*)  (ws + off); off = align256(off + (size_t)NP_ * 3 * 4);
  float*    wts  = (float*)(ws + off); off = align256(off + (size_t)NP_ * 3 * 4);
  _Float16* npts = (_Float16*)(ws + off); off = align256(off + (size_t)NP_ * CIN_ * 2);
  _Float16* x1h  = (_Float16*)(ws + off); off = align256(off + (size_t)NP_ * C1_ * 2);
  float*    Y    = (float*)(ws + off);    // reused: [NP][256] then [NP][128]
  (void)ws_size; (void)in_sizes; (void)n_in; (void)out_size;

  float* stats1 = stats;            // 2*256 floats
  float* stats2 = stats + 2 * 256;  // 2*128 floats (within the 4KB region)

  // 0) weights -> f16 ; zero the stats accumulators
  cvt_weights<<<(C1_ * CIN_ + 255) / 256, 256, 0, stream>>>(w1, w2, w1h, w2h);
  (void)hipMemsetAsync((char*)d_ws + stats_off, 0, 4 * 256 * sizeof(float), stream);

  // 1) 3-NN
  knn3<<<dim3(N_ / 256, B_), 256, 0, stream>>>(xyz1, xyz2, idx, wts);

  // 2) interpolate + concat -> f16 activations [NP][384]
  build_npts<<<(int)(NP_ / 64), 256, 0, stream>>>(points1, points2, idx, wts, npts);

  // 3) layer 1: WMMA GEMM [256x384] x [384xNP] -> Y [NP][256]
  gemm_wmma<C1_, CIN_><<<(int)(NP_ / 64), 256, 0, stream>>>(w1h, npts, Y);
  col_stats<C1_><<<(int)(NP_ / 128), 256, 0, stream>>>(Y, stats1);
  bn_finalize<C1_><<<1, C1_, 0, stream>>>(stats1, g1, beta1, sc1, sh1);
  bn_relu_h<<<(int)(NP_ * C1_ / 256), 256, 0, stream>>>(Y, sc1, sh1, x1h);

  // 4) layer 2: WMMA GEMM [128x256] x [256xNP] -> Y [NP][128] (buffer reused)
  gemm_wmma<C2_, C1_><<<(int)(NP_ / 64), 256, 0, stream>>>(w2h, x1h, Y);
  col_stats<C2_><<<(int)(NP_ / 128), 256, 0, stream>>>(Y, stats2);
  bn_finalize<C2_><<<1, C2_, 0, stream>>>(stats2, g2, beta2, sc2, sh2);

  // 5) BN + ReLU + transpose to [B][128][N]
  bn_relu_out<<<(int)(NP_ / 64), 256, 0, stream>>>(Y, sc2, sh2, out);
}
